// Llama38BDecodeAttention_12962211299335
// MI455X (gfx1250) — matrix operand order repaired and checked by reading, compiled, not run on hardware
//
#include <hip/hip_runtime.h>
#include <hip/hip_bf16.h>
#include <math.h>

typedef __attribute__((ext_vector_type(16))) _Float16 v16h;
typedef __attribute__((ext_vector_type(8)))  float    v8f;

#define N_KV     8
#define N_REP    4
#define HEAD_DIM 128
#define L_CACHE  1024
#define HDIM     4096

// K-index permutation so the 16x32 f16 A-fragment of each lane is one
// contiguous 32B run in LDS (bits [4:3] of k swapped).
__device__ __forceinline__ int permK(int k) {
    return (k & ~31) | (k & 7) | ((k & 8) << 1) | ((k & 16) >> 1);
}

// ---------------------------------------------------------------------------
// RMSNorm: one block per batch row.
// ---------------------------------------------------------------------------
__global__ __launch_bounds__(256) void rmsnorm_kernel(
    const float* __restrict__ x, const float* __restrict__ w,
    float* __restrict__ xn, int H)
{
    __shared__ float red[256];
    const int b = blockIdx.x, tid = threadIdx.x;
    const float* xb = x + (size_t)b * H;
    float s = 0.f;
    for (int i = tid; i < H; i += 256) { float v = xb[i]; s += v * v; }
    red[tid] = s; __syncthreads();
    for (int off = 128; off > 0; off >>= 1) {
        if (tid < off) red[tid] += red[tid + off];
        __syncthreads();
    }
    const float inv = rsqrtf(red[0] / (float)H + 1e-5f);
    for (int i = tid; i < H; i += 256)
        xn[(size_t)b * H + i] = xb[i] * inv * w[i];
}

// ---------------------------------------------------------------------------
// WMMA GEMM: C[128 x ND] = A[128 x KD] * B[KD x ND], compile-time dims.
// Double-buffered LDS tiles staged in WMMA fragment order (A row-major,
// B column-major) -> every fragment is one contiguous v16h (2x ds_load_b128).
// ---------------------------------------------------------------------------
#define BM 128
#define BN 64
#define BK 32

template<int KD, int ND>
__global__ __launch_bounds__(256) void gemm_wmma_kernel(
    const float* __restrict__ A, const float* __restrict__ B,
    float* __restrict__ C)
{
    __shared__ __align__(16) _Float16 As[2][BM * BK];   // 2 x 8 KB
    __shared__ __align__(16) _Float16 Bs[2][BN * BK];   // 2 x 4 KB
    const int tid  = threadIdx.x;
    const int wave = tid >> 5;
    const int lane = tid & 31;
    const int lrow = lane & 15;
    const int lhi  = lane >> 4;
    const int n0   = blockIdx.x * BN;

    auto stage = [&](int k0, int buf) {
        #pragma unroll
        for (int i = 0; i < 4; ++i) {                 // A: 1024 float4s
            int e4 = tid + 256 * i;
            int r = e4 >> 3, c4 = e4 & 7;
            float4 v = *(const float4*)(A + (size_t)r * KD + k0 + c4 * 4);
            int p = r * BK + c4 * 4;
            As[buf][p + 0] = (_Float16)v.x; As[buf][p + 1] = (_Float16)v.y;
            As[buf][p + 2] = (_Float16)v.z; As[buf][p + 3] = (_Float16)v.w;
        }
        #pragma unroll
        for (int i = 0; i < 2; ++i) {                 // B: 512 float4s
            int e4 = tid + 256 * i;
            int r = e4 >> 4, c4 = e4 & 15;
            float4 v = *(const float4*)(B + (size_t)(k0 + r) * ND + n0 + c4 * 4);
            Bs[buf][(c4 * 4 + 0) * BK + r] = (_Float16)v.x;
            Bs[buf][(c4 * 4 + 1) * BK + r] = (_Float16)v.y;
            Bs[buf][(c4 * 4 + 2) * BK + r] = (_Float16)v.z;
            Bs[buf][(c4 * 4 + 3) * BK + r] = (_Float16)v.w;
        }
    };

    v8f acc[4] = {};
    stage(0, 0);
    __syncthreads();

    #pragma unroll 1
    for (int k0 = 0; k0 < KD; k0 += 2 * BK) {
        // buffer 0 compute, prefetch k0+BK into buffer 1
        stage(k0 + BK, 1);
        {
            const v16h af = *(const v16h*)&As[0][(wave * 16 + lrow) * BK + lhi * 16];
            #pragma unroll
            for (int nt = 0; nt < 4; ++nt) {
                const v16h bf = *(const v16h*)&Bs[0][(nt * 16 + lrow) * BK + lhi * 16];
                acc[nt] = __builtin_amdgcn_wmma_f32_16x16x32_f16(
                    false, af, false, bf, (short)0, acc[nt], false, false);
            }
        }
        __syncthreads();
        // buffer 1 compute, prefetch k0+2*BK into buffer 0
        if (k0 + 2 * BK < KD) stage(k0 + 2 * BK, 0);
        {
            const v16h af = *(const v16h*)&As[1][(wave * 16 + lrow) * BK + lhi * 16];
            #pragma unroll
            for (int nt = 0; nt < 4; ++nt) {
                const v16h bf = *(const v16h*)&Bs[1][(nt * 16 + lrow) * BK + lhi * 16];
                acc[nt] = __builtin_amdgcn_wmma_f32_16x16x32_f16(
                    false, af, false, bf, (short)0, acc[nt], false, false);
            }
        }
        __syncthreads();
    }

    // Epilogue: single base address, all offsets compile-time immediates.
    float* Cp = C + (size_t)(wave * 16 + lhi * 8) * ND + n0 + lrow;
    #pragma unroll
    for (int nt = 0; nt < 4; ++nt)
        #pragma unroll
        for (int i = 0; i < 8; ++i)
            Cp[(size_t)i * ND + nt * 16] = acc[nt][i];
}

// ---------------------------------------------------------------------------
// Simplified RoPE of the reference: x *= (cos +/- sin), per head_dim position.
// ---------------------------------------------------------------------------
__global__ void rope_kernel(float* __restrict__ buf, int n, float cacheLen)
{
    int idx = blockIdx.x * 256 + threadIdx.x;
    if (idx >= n) return;
    int d = idx & (HEAD_DIM - 1);
    int i = d & 63;
    float freq = __expf(-(float)i * (13.122363377404328f / 128.0f)); // ln(5e5)
    float ang  = cacheLen * freq;
    float cv = __cosf(ang), sv = __sinf(ang);
    float rope = (d < 64) ? (cv - sv) : (cv + sv);
    buf[idx] *= rope;
}

// ---------------------------------------------------------------------------
// Fused GQA decode attention: one block per (batch, kv-head).
// ---------------------------------------------------------------------------
__global__ __launch_bounds__(256) void attn_kernel(
    const float* __restrict__ q,     // [B, 4096] (rope applied)
    const float* __restrict__ kc,    // [B, 8, 1024, 128]
    const float* __restrict__ vc,    // [B, 8, 1024, 128]
    const float* __restrict__ vnew,  // [B, 8, 128]
    float* __restrict__ out)         // [B, 4096]
{
    __shared__ __align__(16) _Float16 qa[16 * HEAD_DIM];     // row-major, 4 KB
    __shared__ float                  sc[4 * L_CACHE];       // scores, 16 KB
    __shared__ __align__(16) _Float16 pr[16 * L_CACHE];      // probs row-major, 32 KB
    __shared__ __align__(16) _Float16 Vs[2][HEAD_DIM * 64];  // V^T tiles, 2x16 KB
    __shared__ float                  red[4 * 64];

    const int bg = blockIdx.x;
    const int b = bg >> 3, g = bg & 7;
    const int tid = threadIdx.x, wave = tid >> 5, lane = tid & 31;
    const int lrow = lane & 15, lhi = lane >> 4;
    const float scale = 0.08838834764831845f;  // 1/sqrt(128)

    // q (4x128) * scale into row-major layout, zero-pad rows 4-15.
    for (int e = tid; e < 16 * HEAD_DIM; e += 256) {
        int r = e >> 7, d = e & 127;
        _Float16 v = (r < 4)
            ? (_Float16)(q[(size_t)b * HDIM + (g * N_REP + r) * HEAD_DIM + d] * scale)
            : (_Float16)0.f;
        qa[r * HEAD_DIM + d] = v;
    }
    __syncthreads();

    v16h af[4];
    #pragma unroll
    for (int kk = 0; kk < 4; ++kk)
        af[kk] = *(const v16h*)&qa[lrow * HEAD_DIM + kk * 32 + lhi * 16];

    // ---- scores = q @ K^T : 64 tiles of 16 cache rows, 8 per wave ----
    const float* Kb = kc + (size_t)bg * L_CACHE * HEAD_DIM;
    for (int tt = 0; tt < 8; ++tt) {
        const int nt = wave * 8 + tt;
        const float* Krow = Kb + (size_t)(nt * 16 + lrow) * HEAD_DIM;
        __builtin_prefetch((const void*)(Krow + 16 * HEAD_DIM), 0, 0);
        v8f acc = {};
        #pragma unroll
        for (int kk = 0; kk < 4; ++kk) {
            v16h bf;   // B fragment order = sequential K -> float4 loads
            const float4* p = (const float4*)(Krow + 32 * kk + lhi * 16);
            #pragma unroll
            for (int j = 0; j < 4; ++j) {
                float4 v = p[j];
                bf[4 * j + 0] = (_Float16)v.x; bf[4 * j + 1] = (_Float16)v.y;
                bf[4 * j + 2] = (_Float16)v.z; bf[4 * j + 3] = (_Float16)v.w;
            }
            acc = __builtin_amdgcn_wmma_f32_16x16x32_f16(
                false, af[kk], false, bf, (short)0, acc, false, false);
        }
        if (lhi == 0) {            // only lanes 0-15 hold valid rows (m = i < 4)
            #pragma unroll
            for (int i = 0; i < 4; ++i)
                sc[i * L_CACHE + nt * 16 + lrow] = acc[i];
        }
    }
    __syncthreads();

    // ---- softmax over L=1024: 4 rows x 64 threads ----
    {
        const int r = tid >> 6, c = tid & 63;
        float pm = -1e30f;
        #pragma unroll 4
        for (int i = 0; i < 16; ++i) pm = fmaxf(pm, sc[r * L_CACHE + c * 16 + i]);
        red[r * 64 + c] = pm; __syncthreads();
        for (int off = 32; off > 0; off >>= 1) {
            if (c < off) red[r * 64 + c] = fmaxf(red[r * 64 + c], red[r * 64 + c + off]);
            __syncthreads();
        }
        const float rowmax = red[r * 64];
        __syncthreads();
        float ps = 0.f;
        #pragma unroll 4
        for (int i = 0; i < 16; ++i) {
            float e = __expf(sc[r * L_CACHE + c * 16 + i] - rowmax);
            sc[r * L_CACHE + c * 16 + i] = e;
            ps += e;
        }
        red[r * 64 + c] = ps; __syncthreads();
        for (int off = 32; off > 0; off >>= 1) {
            if (c < off) red[r * 64 + c] += red[r * 64 + c + off];
            __syncthreads();
        }
        const float inv = 1.f / red[r * 64];
        #pragma unroll 4
        for (int i = 0; i < 16; ++i) {
            int l = c * 16 + i;
            pr[r * L_CACHE + l] = (_Float16)(sc[r * L_CACHE + l] * inv);
        }
        for (int e = tid; e < 12 * L_CACHE; e += 256)   // zero rows 4-15
            pr[4 * L_CACHE + e] = (_Float16)0.f;
    }
    __syncthreads();

    // ---- out = probs @ V : double-buffered V^T staging through LDS ----
    const float* Vb = vc + (size_t)bg * L_CACHE * HEAD_DIM;
    const int dcol = wave * 16 + lrow;

    auto stageV = [&](int s, int buf) {
        #pragma unroll
        for (int i = 0; i < 8; ++i) {          // 64 rows x 128 cols = 2048 float4
            int e4 = tid + 256 * i;
            int r = e4 >> 5, c4 = e4 & 31;
            float4 v = *(const float4*)(Vb + (size_t)(s * 64 + r) * HEAD_DIM + c4 * 4);
            Vs[buf][(c4 * 4 + 0) * 64 + r] = (_Float16)v.x;
            Vs[buf][(c4 * 4 + 1) * 64 + r] = (_Float16)v.y;
            Vs[buf][(c4 * 4 + 2) * 64 + r] = (_Float16)v.z;
            Vs[buf][(c4 * 4 + 3) * 64 + r] = (_Float16)v.w;
        }
    };
    auto tileV = [&](int s, int buf, v8f& oa) {
        #pragma unroll
        for (int ss = 0; ss < 2; ++ss) {
            int kk = s * 2 + ss;
            const v16h pa = *(const v16h*)&pr[lrow * L_CACHE + kk * 32 + lhi * 16];
            const v16h bf = *(const v16h*)&Vs[buf][dcol * 64 + ss * 32 + lhi * 16];
            oa = __builtin_amdgcn_wmma_f32_16x16x32_f16(
                false, pa, false, bf, (short)0, oa, false, false);
        }
    };

    v8f oacc = {};
    stageV(0, 0);
    __syncthreads();
    #pragma unroll 1
    for (int s = 0; s < 16; s += 2) {
        stageV(s + 1, 1);              // overlap loads with buffer-0 WMMAs
        tileV(s, 0, oacc);
        __syncthreads();
        if (s + 2 < 16) stageV(s + 2, 0);
        tileV(s + 1, 1, oacc);
        __syncthreads();
    }

    if (lhi == 0) {                    // rows m = 0..3 live in lanes 0-15
        float* op = out + (size_t)b * HDIM + (g * N_REP) * HEAD_DIM + dcol;
        const float vn = vnew[(size_t)bg * HEAD_DIM + dcol];
        #pragma unroll
        for (int i = 0; i < 4; ++i)
            op[i * HEAD_DIM] = oacc[i] + vn;
    }
}

// ---------------------------------------------------------------------------
extern "C" void kernel_launch(void* const* d_in, const int* in_sizes, int n_in,
                              void* d_out, int out_size, void* d_ws, size_t ws_size,
                              hipStream_t stream)
{
    const float* x   = (const float*)d_in[0];
    const float* kc  = (const float*)d_in[1];
    const float* vc  = (const float*)d_in[2];
    const float* nw  = (const float*)d_in[3];
    const float* wq  = (const float*)d_in[4];
    // d_in[5] = wk is provably dead: the reference's new-token softmax is over
    // a length-1 axis, so its probability is identically 1.0.
    const float* wv  = (const float*)d_in[6];
    const float* wo  = (const float*)d_in[7];
    float* out = (float*)d_out;

    const int H = in_sizes[3];                           // 4096
    const int B = in_sizes[0] / H;                       // 128
    const int L = in_sizes[1] / (B * N_KV * HEAD_DIM);   // 1024

    float* xn   = (float*)d_ws;                          // [B, H]
    float* qb   = xn   + (size_t)B * H;                  // [B, 4096]
    float* vnew = qb   + (size_t)B * H;                  // [B, 8*128]
    float* attn = vnew + (size_t)B * N_KV * HEAD_DIM;    // [B, 4096]

    rmsnorm_kernel<<<B, 256, 0, stream>>>(x, nw, xn, H);

    gemm_wmma_kernel<HDIM, HDIM><<<dim3(HDIM / BN), 256, 0, stream>>>(xn, wq, qb);
    gemm_wmma_kernel<HDIM, N_KV * HEAD_DIM>
        <<<dim3((N_KV * HEAD_DIM) / BN), 256, 0, stream>>>(xn, wv, vnew);

    {
        int n = B * HDIM;
        rope_kernel<<<(n + 255) / 256, 256, 0, stream>>>(qb, n, (float)L);
    }

    attn_kernel<<<B * N_KV, 256, 0, stream>>>(qb, kc, vc, vnew, attn);

    gemm_wmma_kernel<HDIM, HDIM><<<dim3(HDIM / BN), 256, 0, stream>>>(attn, wo, out);
}
